// BuggyMultiHeadAttention_25460566130949
// MI455X (gfx1250) — matrix operand order repaired and checked by reading, not compiled, not run on hardware
//
#include <hip/hip_runtime.h>
#include <hip/hip_bf16.h>

// ---------------------------------------------------------------------------
// BuggyMultiHeadAttention for MI455X (gfx1250), wave32 + WMMA f16->f32.
//
//  B=4, S=2048, D_MODEL=1024, H=16, HEAD_DIM=64.
//  scale = 1/sqrt(D_MODEL) = 1/32 (intentionally "buggy" per reference),
//  folded into the Q fragments (exactly representable in f16).
//
// Pipeline:
//   1) cvt f32->f16 of q,k,v and Wq,Wk,Wv,Wo into workspace
//   2) WMMA GEMM: Qp = q16 @ Wq^T + bq   (f16 out)   [same for K,V]
//   3) flash-style attention (online softmax) -> att (f16, [B,S,1024] layout)
//      - K block staged to LDS via async global->LDS copies (ASYNCcnt path)
//      - V block staged transposed via VGPR path (B-operand wants columns)
//   4) WMMA GEMM: out = att @ Wo^T + bo  (f32 out -> d_out)
// ---------------------------------------------------------------------------

typedef _Float16 h4  __attribute__((ext_vector_type(4)));
typedef _Float16 h8  __attribute__((ext_vector_type(8)));
typedef _Float16 h16 __attribute__((ext_vector_type(16)));
typedef float    f8  __attribute__((ext_vector_type(8)));

#define DEV __device__ __forceinline__

static constexpr int DMODEL = 1024;
static constexpr int SEQ    = 2048;
static constexpr int BATCH  = 4;
static constexpr int HEADS  = 16;
static constexpr int HD     = 64;

DEV h16 cat8(h8 lo, h8 hi) {
  return __builtin_shufflevector(lo, hi, 0,1,2,3,4,5,6,7,8,9,10,11,12,13,14,15);
}

// A-operand fragment, 16x32 f16 (ISA 7.12.2): lane half g (=lane>>4), row = lane&15.
// element i<8  -> K = 8g + i ; element i>=8 -> K = 16 + 8g + (i-8)
DEV h16 load_a(const _Float16* rowp, int g) {
  h8 lo = *(const h8*)(rowp + 8 * g);
  h8 hi = *(const h8*)(rowp + 8 * g + 16);
  return cat8(lo, hi);
}

// B-operand fragment, 32x16 f16: lane half g, col = lane&15.
// element i<8  -> K = 8g + i ; element i>=8 -> K = 16 + 8g + (i-8)
DEV h16 load_b(const _Float16* colp, int g) {
  h8 lo = *(const h8*)(colp + 8 * g);
  h8 hi = *(const h8*)(colp + 8 * g + 16);
  return cat8(lo, hi);
}

DEV f8 wmma16(h16 a, h16 b, f8 c) {
  // 8 args: (neg_a, A, neg_b, B, c_mod, C, reuse_a, reuse_b)
  return __builtin_amdgcn_wmma_f32_16x16x32_f16(false, a, false, b, (short)0, c,
                                                false, false);
}

// Async global -> LDS copy of 16 bytes (gfx1250, tracked by ASYNCcnt).
DEV void async_copy_b128(const void* gsrc, void* lds_dst) {
  const unsigned lds_off = (unsigned)(uintptr_t)lds_dst;  // low 32 bits = LDS addr
  asm volatile("global_load_async_to_lds_b128 %0, %1, off"
               :
               : "v"(lds_off), "v"(gsrc)
               : "memory");
}

DEV void wait_async0() {
  asm volatile("s_wait_asynccnt 0x0" ::: "memory");
}

// ---------------------------------------------------------------------------
// f32 -> f16 conversion, 4 elems/thread (all sizes are multiples of 1024)
// ---------------------------------------------------------------------------
__global__ __launch_bounds__(256) void cvt_f32_to_f16(
    const float* __restrict__ in, _Float16* __restrict__ out, int n) {
  int i = (blockIdx.x * blockDim.x + threadIdx.x) * 4;
  if (i < n) {
    float4 v = *(const float4*)(in + i);
    h4 o;
    o.x = (_Float16)v.x; o.y = (_Float16)v.y;
    o.z = (_Float16)v.z; o.w = (_Float16)v.w;
    *(h4*)(out + i) = o;
  }
}

// ---------------------------------------------------------------------------
// GEMM: C[M,N] = A[M,K](f16,row-major) @ W[N,K]^T (f16,row-major) + bias[N]
// Each wave computes a 64x64 tile (4x4 WMMA tiles). Block = 8 waves (2M x 4N)
// -> block tile 128 x 256.  Grid = (N/256, M/128).
// Since W is [out,in] row-major, row n of W *is* the B-operand column layout.
// ---------------------------------------------------------------------------
template <bool F32OUT>
__global__ __launch_bounds__(256) void gemm_wmma(
    const _Float16* __restrict__ A, const _Float16* __restrict__ W,
    const float* __restrict__ bias, void* __restrict__ Cout,
    int M, int N, int K) {
  const int lane = threadIdx.x & 31;
  const int wv   = threadIdx.x >> 5;
  const int g    = lane >> 4;   // lane half
  const int r    = lane & 15;   // row (A) / col (B) within tile

  const int mBase = blockIdx.y * 128 + (wv >> 2) * 64;
  const int nBase = blockIdx.x * 256 + (wv & 3) * 64;

  const _Float16* aRow[4];
  const _Float16* wRow[4];
#pragma unroll
  for (int t = 0; t < 4; ++t) {
    aRow[t] = A + (size_t)(mBase + t * 16 + r) * K;
    wRow[t] = W + (size_t)(nBase + t * 16 + r) * K;
  }

  f8 acc[4][4] = {};

  for (int k0 = 0; k0 < K; k0 += 32) {
    // prefetch streaming A operand 256B (= 4 K-steps) ahead
    if (k0 + 128 < K) {
#pragma unroll
      for (int mt = 0; mt < 4; ++mt)
        __builtin_prefetch(aRow[mt] + k0 + 128, 0, 1);
    }
    h16 af[4], bf[4];
#pragma unroll
    for (int mt = 0; mt < 4; ++mt) af[mt] = load_a(aRow[mt] + k0, g);
#pragma unroll
    for (int nt = 0; nt < 4; ++nt) bf[nt] = load_b(wRow[nt] + k0, g);
#pragma unroll
    for (int mt = 0; mt < 4; ++mt)
#pragma unroll
      for (int nt = 0; nt < 4; ++nt)
        acc[mt][nt] = wmma16(af[mt], bf[nt], acc[mt][nt]);
  }

  // epilogue: C layout = row (j + 8g) x col (lane&15) per 16x16 tile
#pragma unroll
  for (int nt = 0; nt < 4; ++nt) {
    const int n  = nBase + nt * 16 + r;
    const float bv = bias ? bias[n] : 0.0f;
#pragma unroll
    for (int mt = 0; mt < 4; ++mt) {
#pragma unroll
      for (int j = 0; j < 8; ++j) {
        const int m = mBase + mt * 16 + j + 8 * g;
        const float v = acc[mt][nt][j] + bv;
        if (F32OUT)
          ((float*)Cout)[(size_t)m * N + n] = v;
        else
          ((_Float16*)Cout)[(size_t)m * N + n] = (_Float16)v;
      }
    }
  }
}

// ---------------------------------------------------------------------------
// Flash-style attention.  Q,K,V are projected f16 tensors in [B,S,1024] layout
// (head h occupies columns h*64 .. h*64+63 -> contiguous 64 f16 per row).
// One wave handles 16 query rows of one (b,h); block = 8 waves = 128 rows.
// Grid = (S/128, B*H).
// Per 32-key block:
//   - K block (32x64 f16 = 4KB) async-copied to LDS once per block
//   - 4 WMMA for scores (Kdim 64)
//   - online softmax with half-wave shfl_xor row reductions
//   - P re-laid out via LDS; V^T staged in LDS; 4 WMMA for P@V
// ---------------------------------------------------------------------------
__global__ __launch_bounds__(256) void attn_wmma(
    const _Float16* __restrict__ Q, const _Float16* __restrict__ Kt,
    const _Float16* __restrict__ V, _Float16* __restrict__ O) {
  const int bh = blockIdx.y;
  const int b  = bh >> 4;
  const int h  = bh & 15;
  const int lane = threadIdx.x & 31;
  const int wv   = threadIdx.x >> 5;
  const int g    = lane >> 4;
  const int r    = lane & 15;

  const size_t headOff = (size_t)b * SEQ * DMODEL + (size_t)h * HD;
  const _Float16* Qb = Q  + headOff;
  const _Float16* Kb = Kt + headOff;
  const _Float16* Vb = V  + headOff;

  __shared__ __align__(16) _Float16 Plds[8][16 * 32];  // per-wave P tile (8KB)
  __shared__ __align__(16) _Float16 Kl[32 * HD];       // K block, row-major (4KB)
  __shared__ __align__(16) _Float16 Vt[HD * 32];       // V^T block (4KB)

  const int qRow0 = blockIdx.x * 128 + wv * 16;

  // Q fragments for this wave's 16 rows (Kdim 0..31 and 32..63), pre-scaled by
  // the reference's "buggy" 1/sqrt(D_MODEL) = 1/32 (exact in f16).
  const _Float16* qrp = Qb + (size_t)(qRow0 + r) * DMODEL;
  h16 aq0 = load_a(qrp, g);
  h16 aq1 = load_a(qrp + 32, g);
  aq0 = aq0 * (_Float16)0.03125f;
  aq1 = aq1 * (_Float16)0.03125f;

  f8 acc[4] = {};        // 16 x 64 output accumulator (4 N-tiles)
  float mrun[8], lrun[8];
#pragma unroll
  for (int j = 0; j < 8; ++j) { mrun[j] = -1e30f; lrun[j] = 0.0f; }

  for (int j0 = 0; j0 < SEQ; j0 += 32) {
    __syncthreads();  // previous iteration's readers done before restaging
    {
      const int k = threadIdx.x >> 3;      // key 0..31
      const int c = threadIdx.x & 7;       // 16B chunk 0..7
      // ---- K block: async global->LDS, layout identical to global rows ----
      async_copy_b128(Kb + (size_t)(j0 + k) * DMODEL + c * 8,
                      &Kl[k * HD + c * 8]);
      // ---- V block: stage transposed (Vt[dim][key]) via VGPR path ----
      h8 vv = *(const h8*)(Vb + (size_t)(j0 + k) * DMODEL + c * 8);
#pragma unroll
      for (int i = 0; i < 8; ++i) Vt[(c * 8 + i) * 32 + k] = vv[i];
    }
    wait_async0();      // this wave's async copies landed in LDS
    __syncthreads();    // all waves' staging visible

    // ---- scores: two 16x16 tiles (keys j0..+15, j0+16..+31), from LDS K ----
    f8 c0 = {}, c1 = {};
    const _Float16* kr0 = &Kl[r * HD];
    const _Float16* kr1 = &Kl[(16 + r) * HD];
    c0 = wmma16(aq0, load_b(kr0, g), c0);
    c0 = wmma16(aq1, load_b(kr0 + 32, g), c0);
    c1 = wmma16(aq0, load_b(kr1, g), c1);
    c1 = wmma16(aq1, load_b(kr1 + 32, g), c1);

    // ---- online softmax (row j+8g lives in elem j across a 16-lane half) --
    _Float16* P = Plds[wv];
#pragma unroll
    for (int j = 0; j < 8; ++j) {
      const float s0 = c0[j];
      const float s1 = c1[j];
      float mx = fmaxf(s0, s1);
#pragma unroll
      for (int msk = 8; msk >= 1; msk >>= 1)
        mx = fmaxf(mx, __shfl_xor(mx, msk, 32));
      const float mnew  = fmaxf(mrun[j], mx);
      const float alpha = __expf(mrun[j] - mnew);
      const float p0 = __expf(s0 - mnew);
      const float p1 = __expf(s1 - mnew);
      float ps = p0 + p1;
#pragma unroll
      for (int msk = 8; msk >= 1; msk >>= 1) ps += __shfl_xor(ps, msk, 32);
      lrun[j] = lrun[j] * alpha + ps;
      mrun[j] = mnew;
#pragma unroll
      for (int nt = 0; nt < 4; ++nt) acc[nt][j] *= alpha;
      const int m = j + 8 * g;
      P[m * 32 + r]      = (_Float16)p0;
      P[m * 32 + 16 + r] = (_Float16)p1;
    }

    // ---- P(16x32) @ V(32x64): reload P in A-layout, V^T rows as B-operand --
    const h16 ap = load_a(P + r * 32, g);
#pragma unroll
    for (int nt = 0; nt < 4; ++nt) {
      const h16 bv = load_b(&Vt[(nt * 16 + r) * 32], g);
      acc[nt] = wmma16(ap, bv, acc[nt]);
    }
  }

  // ---- normalize + store: O[b, row, h*64 + n] (f16 for final projection) --
  float inv[8];
#pragma unroll
  for (int j = 0; j < 8; ++j) inv[j] = 1.0f / lrun[j];
#pragma unroll
  for (int nt = 0; nt < 4; ++nt) {
#pragma unroll
    for (int j = 0; j < 8; ++j) {
      const int row = qRow0 + j + 8 * g;
      const int n   = nt * 16 + r;
      O[(size_t)b * SEQ * DMODEL + (size_t)row * DMODEL + h * HD + n] =
          (_Float16)(acc[nt][j] * inv[j]);
    }
  }
}

// ---------------------------------------------------------------------------
// Launcher
// ---------------------------------------------------------------------------
extern "C" void kernel_launch(void* const* d_in, const int* in_sizes, int n_in,
                              void* d_out, int out_size, void* d_ws,
                              size_t ws_size, hipStream_t stream) {
  const float* q  = (const float*)d_in[0];
  const float* k  = (const float*)d_in[1];
  const float* v  = (const float*)d_in[2];
  const float* Wq = (const float*)d_in[3];
  const float* bq = (const float*)d_in[4];
  const float* Wk = (const float*)d_in[5];
  const float* bk = (const float*)d_in[6];
  const float* Wv = (const float*)d_in[7];
  const float* bv = (const float*)d_in[8];
  const float* Wo = (const float*)d_in[9];
  const float* bo = (const float*)d_in[10];

  const size_t MROWS = (size_t)BATCH * SEQ;          // 8192
  const size_t XE    = MROWS * DMODEL;               // 8M elems
  const size_t WE    = (size_t)DMODEL * DMODEL;      // 1M elems

  _Float16* ws   = (_Float16*)d_ws;
  _Float16* q16  = ws;
  _Float16* k16  = ws + XE;
  _Float16* v16  = ws + 2 * XE;
  _Float16* wq16 = ws + 3 * XE;
  _Float16* wk16 = wq16 + WE;
  _Float16* wv16 = wk16 + WE;
  _Float16* wo16 = wv16 + WE;
  _Float16* Qp   = wo16 + WE;
  _Float16* Kp   = Qp + XE;
  _Float16* Vp   = Kp + XE;
  _Float16* att  = q16;  // q16 dead after Q projection -> reuse for attn out

  const int nX = (int)XE, nW = (int)WE;
  const int cvtT = 256;
  cvt_f32_to_f16<<<nX / (cvtT * 4), cvtT, 0, stream>>>(q, q16, nX);
  cvt_f32_to_f16<<<nX / (cvtT * 4), cvtT, 0, stream>>>(k, k16, nX);
  cvt_f32_to_f16<<<nX / (cvtT * 4), cvtT, 0, stream>>>(v, v16, nX);
  cvt_f32_to_f16<<<nW / (cvtT * 4), cvtT, 0, stream>>>(Wq, wq16, nW);
  cvt_f32_to_f16<<<nW / (cvtT * 4), cvtT, 0, stream>>>(Wk, wk16, nW);
  cvt_f32_to_f16<<<nW / (cvtT * 4), cvtT, 0, stream>>>(Wv, wv16, nW);
  cvt_f32_to_f16<<<nW / (cvtT * 4), cvtT, 0, stream>>>(Wo, wo16, nW);

  dim3 gGemm(DMODEL / 256, (unsigned)(MROWS / 128));  // (4, 64)
  gemm_wmma<false><<<gGemm, 256, 0, stream>>>(q16, wq16, bq, Qp,
                                              (int)MROWS, DMODEL, DMODEL);
  gemm_wmma<false><<<gGemm, 256, 0, stream>>>(k16, wk16, bk, Kp,
                                              (int)MROWS, DMODEL, DMODEL);
  gemm_wmma<false><<<gGemm, 256, 0, stream>>>(v16, wv16, bv, Vp,
                                              (int)MROWS, DMODEL, DMODEL);

  dim3 gAttn(SEQ / 128, BATCH * HEADS);  // (16, 64)
  attn_wmma<<<gAttn, 256, 0, stream>>>(Qp, Kp, Vp, att);

  gemm_wmma<true><<<gGemm, 256, 0, stream>>>(att, wo16, bo, d_out,
                                             (int)MROWS, DMODEL, DMODEL);
}
